// MultiHeadAttentionLayer_16183436771963
// MI455X (gfx1250) — matrix operand order repaired and checked by reading, not compiled, not run on hardware
//
#include <hip/hip_runtime.h>

// ---------------------------------------------------------------------------
// MI455X (gfx1250) relative-position MHA, bf16 WMMA (v_wmma_f32_16x16x32_bf16)
// B=2, S=2048, HID=1024, NH=16, D=64, MAXREL=4
// ---------------------------------------------------------------------------

typedef __bf16 bf16_t;
typedef __attribute__((ext_vector_type(16))) __bf16 bf16x16;
typedef __attribute__((ext_vector_type(8)))  __bf16 bf16x8;
typedef __attribute__((ext_vector_type(8)))  float  f32x8;

#define S_LEN 2048
#define NHEAD 16
#define DHEAD 64
#define HIDN  1024
#define BATCH 2
#define MROWS (BATCH * S_LEN)   // 4096

// ---- WMMA wrapper ----------------------------------------------------------
__device__ __forceinline__ f32x8 wmma_bf16(bf16x16 a, bf16x16 b, f32x8 c) {
  return __builtin_amdgcn_wmma_f32_16x16x32_bf16(false, a, false, b,
                                                 (short)0, c, false, false);
}

// A-frag (16x32 bf16, row-major source, lda elements):
// lane: m = lane&15, half = lane>>4 ; element j -> K = 16*(j>>3)+8*half+(j&7)
__device__ __forceinline__ bf16x16 load_a_frag(const bf16_t* A, int lda,
                                               int row0, int k0) {
  const int lane = threadIdx.x & 31;
  const int m = lane & 15, hf = lane >> 4;
  const bf16_t* p = A + (size_t)(row0 + m) * lda + k0 + 8 * hf;
  bf16x8 lo = *(const bf16x8*)p;
  bf16x8 hi = *(const bf16x8*)(p + 16);
  bf16x16 r;
#pragma unroll
  for (int j = 0; j < 8; ++j) { r[j] = lo[j]; r[j + 8] = hi[j]; }
  return r;
}

// B-frag (32x16 bf16). Source is "BT" = column-per-row layout [N][Kdim]:
// lane: n = lane&15, half = lane>>4 ; element j -> K = 16*(j>>3)+8*half+(j&7)
__device__ __forceinline__ bf16x16 load_b_frag(const bf16_t* BT, int ldb,
                                               int n0, int k0) {
  const int lane = threadIdx.x & 31;
  const int n = lane & 15, hf = lane >> 4;
  const bf16_t* p = BT + (size_t)(n0 + n) * ldb + k0 + 8 * hf;
  bf16x8 lo = *(const bf16x8*)p;
  bf16x8 hi = *(const bf16x8*)(p + 16);
  bf16x16 r;
#pragma unroll
  for (int j = 0; j < 8; ++j) { r[j] = lo[j]; r[j + 8] = hi[j]; }
  return r;
}

// ---- trivial prep kernels --------------------------------------------------
__global__ void cast_f32_to_bf16(const float* __restrict__ in,
                                 bf16_t* __restrict__ out, int nelem) {
  int i = blockIdx.x * blockDim.x + threadIdx.x;
  if (i < nelem) out[i] = (bf16_t)in[i];
}

// W [K][N] row-major -> WT [N][K] bf16
__global__ void transpose_cast(const float* __restrict__ W,
                               bf16_t* __restrict__ WT) {
  int i = blockIdx.x * blockDim.x + threadIdx.x;
  if (i >= HIDN * HIDN) return;
  int k = i & (HIDN - 1), n = i >> 10;
  WT[i] = (bf16_t)W[(size_t)k * HIDN + n];
}

// qrel[bh][s][r] = dot(Q[bh][s][:], table_k[r][:])
__global__ void qrel_kernel(const bf16_t* __restrict__ Q,
                            const float* __restrict__ table_k,
                            float* __restrict__ qrel) {
  const int total = BATCH * NHEAD * S_LEN * 9;
  int i = blockIdx.x * blockDim.x + threadIdx.x;
  if (i >= total) return;
  int r = i % 9;
  int qs = i / 9;
  const bf16_t* qp = Q + (size_t)qs * DHEAD;
  const float* t = table_k + r * DHEAD;
  float a = 0.f;
#pragma unroll
  for (int d = 0; d < DHEAD; ++d) a += (float)qp[d] * t[d];
  qrel[i] = a;
}

// ---- GEMM: C = A(bf16 [M][K]) @ BT(bf16 [N][K]) + bias ---------------------
// 32x64 output tile per wave (2048 waves for 4096x1024 -> ~2 waves/SIMD for
// latency hiding; whole working set is L2-resident on the 192MB L2).
// omode 0: bf16 out, head layout  [bh][s][d]      (Q, K)
// omode 1: bf16 out, transposed V [bh][d][s]      (V)
// omode 2: f32  out, row-major    [M][N]          (final)
__global__ __launch_bounds__(128) void gemm_wmma(
    const bf16_t* __restrict__ A, const bf16_t* __restrict__ BT,
    const float* __restrict__ bias, void* __restrict__ out,
    int M, int N, int K, int omode) {
  const int lane = threadIdx.x & 31;
  const int nn = lane & 15, hf = lane >> 4;
  const int gwave = blockIdx.x * (blockDim.x >> 5) + (threadIdx.x >> 5);
  const int tiles_n = N >> 6;
  const int tm = gwave / tiles_n, tn = gwave % tiles_n;
  const int row0 = tm << 5, col0 = tn << 6;
  if (row0 >= M) return;

  const f32x8 vzero = {0.f, 0.f, 0.f, 0.f, 0.f, 0.f, 0.f, 0.f};
  f32x8 acc[2][4];
#pragma unroll
  for (int i = 0; i < 2; ++i)
#pragma unroll
    for (int j = 0; j < 4; ++j) acc[i][j] = vzero;

  for (int k0 = 0; k0 < K; k0 += 32) {
    bf16x16 af[2], bfR[4];
#pragma unroll
    for (int mt = 0; mt < 2; ++mt) af[mt] = load_a_frag(A, K, row0 + mt * 16, k0);
#pragma unroll
    for (int nt = 0; nt < 4; ++nt) bfR[nt] = load_b_frag(BT, K, col0 + nt * 16, k0);
    // speculative prefetch of next k-step (global_prefetch_b8; dropped if OOB)
    __builtin_prefetch(A + (size_t)(row0 + nn) * K + k0 + 32, 0, 1);
    __builtin_prefetch(BT + (size_t)(col0 + (lane & 63)) * K + k0 + 32, 0, 1);
#pragma unroll
    for (int mt = 0; mt < 2; ++mt)
#pragma unroll
      for (int nt = 0; nt < 4; ++nt)
        acc[mt][nt] = wmma_bf16(af[mt], bfR[nt], acc[mt][nt]);
  }

#pragma unroll
  for (int mt = 0; mt < 2; ++mt)
#pragma unroll
    for (int nt = 0; nt < 4; ++nt) {
      int ng = col0 + nt * 16 + nn;
      float bv = bias[ng];
#pragma unroll
      for (int r = 0; r < 8; ++r) {
        int mg = row0 + mt * 16 + 8 * hf + r;          // C row = 8*half + r
        float val = acc[mt][nt][r] + bv;
        if (omode == 2) {
          ((float*)out)[(size_t)mg * N + ng] = val;
        } else {
          int bb = mg >> 11, s = mg & (S_LEN - 1);
          int hh = ng >> 6, d = ng & 63;
          size_t bh = (size_t)(bb * NHEAD + hh);
          if (omode == 0)
            ((bf16_t*)out)[(bh * S_LEN + s) * DHEAD + d] = (bf16_t)val;
          else
            ((bf16_t*)out)[(bh * DHEAD + d) * S_LEN + s] = (bf16_t)val;
        }
      }
    }
}

// ---- fused flash attention with relative position terms --------------------
// grid: BATCH*NHEAD*32 blocks, 128 threads (4 waves). Each wave owns 16 q rows.
__global__ __launch_bounds__(128) void attn_kernel(
    const bf16_t* __restrict__ Q, const bf16_t* __restrict__ Km,
    const bf16_t* __restrict__ Vt, const float* __restrict__ qrel,
    const float* __restrict__ table_v, bf16_t* __restrict__ ctx) {
  __shared__ float tv[9 * DHEAD];
  __shared__ float qr[4][16 * 9];
  __shared__ __align__(16) bf16_t pstage[4][16 * 32];

  const int tid = threadIdx.x;
  const int wave = tid >> 5, lane = tid & 31;
  const int n = lane & 15, hf = lane >> 4;

  const int bh = blockIdx.x >> 5;      // (b*NHEAD + h)
  const int chunk = blockIdx.x & 31;   // 64-row chunk of S
  const int q0 = chunk * 64 + wave * 16;
  const int b = bh >> 4, h = bh & 15;

  const bf16_t* Qb = Q + (size_t)bh * S_LEN * DHEAD;
  const bf16_t* Kb = Km + (size_t)bh * S_LEN * DHEAD;
  const bf16_t* Vb = Vt + (size_t)bh * DHEAD * S_LEN;
  const float* qrb = qrel + (size_t)bh * S_LEN * 9;

  for (int i = tid; i < 9 * DHEAD; i += 128) tv[i] = table_v[i];
  for (int i = lane; i < 16 * 9; i += 32) qr[wave][i] = qrb[q0 * 9 + i];
  __syncthreads();

  const bf16x16 qa0 = load_a_frag(Qb, DHEAD, q0, 0);
  const bf16x16 qa1 = load_a_frag(Qb, DHEAD, q0, 32);

  const f32x8 vzero = {0.f, 0.f, 0.f, 0.f, 0.f, 0.f, 0.f, 0.f};
  f32x8 acc[4];
#pragma unroll
  for (int i = 0; i < 4; ++i) acc[i] = vzero;

  float rmax[8], rsum[8], buck[8][9];
#pragma unroll
  for (int r = 0; r < 8; ++r) {
    rmax[r] = -1e30f;
    rsum[r] = 0.f;
#pragma unroll
    for (int j = 0; j < 9; ++j) buck[r][j] = 0.f;
  }

  for (int kb = 0; kb < S_LEN; kb += 32) {
    // prefetch next key/value block (speculative; dropped if OOB)
    __builtin_prefetch(Kb + (size_t)(kb + 32 + n) * DHEAD, 0, 1);
    __builtin_prefetch(Vb + (size_t)n * S_LEN + kb + 32, 0, 1);
    // ---- scores: 16 q rows x 32 keys, contraction over D=64 (2 k-frags)
    f32x8 sc[2];
    sc[0] = vzero; sc[1] = vzero;
#pragma unroll
    for (int kt = 0; kt < 2; ++kt) {
      bf16x16 a = kt ? qa1 : qa0;
#pragma unroll
      for (int t = 0; t < 2; ++t) {
        bf16x16 kfr = load_b_frag(Kb, DHEAD, kb + t * 16, kt * 32);
        sc[t] = wmma_bf16(a, kfr, sc[t]);
      }
    }
    // ---- add content-position bias and scale by 1/sqrt(64)
    float p[2][8];
#pragma unroll
    for (int t = 0; t < 2; ++t) {
      int c = kb + t * 16 + n;
#pragma unroll
      for (int r = 0; r < 8; ++r) {
        int qrow = 8 * hf + r;
        int delta = c - (q0 + qrow);
        int ri = delta < -4 ? 0 : (delta > 4 ? 8 : delta + 4);
        p[t][r] = (sc[t][r] + qr[wave][qrow * 9 + ri]) * 0.125f;
      }
    }
    // ---- online softmax per row (row lives in 16 lanes of one half-wave)
#pragma unroll
    for (int r = 0; r < 8; ++r) {
      float bm = fmaxf(p[0][r], p[1][r]);
#pragma unroll
      for (int off = 1; off < 16; off <<= 1)
        bm = fmaxf(bm, __shfl_xor(bm, off, 32));
      float nm = fmaxf(rmax[r], bm);
      float sf = __expf(rmax[r] - nm);
      rmax[r] = nm;
      rsum[r] *= sf;
#pragma unroll
      for (int j = 0; j < 9; ++j) buck[r][j] *= sf;
#pragma unroll
      for (int nt = 0; nt < 4; ++nt) acc[nt][r] *= sf;

      float e0 = __expf(p[0][r] - nm);
      float e1 = __expf(p[1][r] - nm);
      p[0][r] = e0; p[1][r] = e1;
      float es = e0 + e1;
#pragma unroll
      for (int off = 1; off < 16; off <<= 1) es += __shfl_xor(es, off, 32);
      rsum[r] += es;

      // per-lane relative-position bucket partials (register resident)
      int qg = q0 + 8 * hf + r;
      int d0 = (kb + n) - qg, d1 = (kb + 16 + n) - qg;
      int r0 = d0 < -4 ? 0 : (d0 > 4 ? 8 : d0 + 4);
      int r1 = d1 < -4 ? 0 : (d1 > 4 ? 8 : d1 + 4);
#pragma unroll
      for (int j = 0; j < 9; ++j)
        buck[r][j] += (j == r0 ? e0 : 0.f) + (j == r1 ? e1 : 0.f);
    }
    // ---- stage probs to LDS (C-layout) and reload as A-frag (same wave,
    //      DS ops are in-order within a wave; compiler barrier stops reorder)
#pragma unroll
    for (int t = 0; t < 2; ++t)
#pragma unroll
      for (int r = 0; r < 8; ++r)
        pstage[wave][(8 * hf + r) * 32 + t * 16 + n] = (bf16_t)p[t][r];
    asm volatile("" ::: "memory");
    bf16x16 pa;
    {
      const bf16_t* base = &pstage[wave][(lane & 15) * 32 + 8 * hf];
      bf16x8 lo = *(const bf16x8*)base;
      bf16x8 hi = *(const bf16x8*)(base + 16);
#pragma unroll
      for (int j = 0; j < 8; ++j) { pa[j] = lo[j]; pa[j + 8] = hi[j]; }
    }
    asm volatile("" ::: "memory");
    // ---- P @ V : contraction over the 32 keys, 4 d-tiles
#pragma unroll
    for (int nt = 0; nt < 4; ++nt) {
      bf16x16 vfr = load_b_frag(Vb, S_LEN, nt * 16, kb);
      acc[nt] = wmma_bf16(pa, vfr, acc[nt]);
    }
  }

  // ---- reduce bucket partials across the 16 lanes holding each row
#pragma unroll
  for (int r = 0; r < 8; ++r)
#pragma unroll
    for (int j = 0; j < 9; ++j) {
      float v = buck[r][j];
#pragma unroll
      for (int off = 1; off < 16; off <<= 1) v += __shfl_xor(v, off, 32);
      buck[r][j] = v;
    }

  // ---- out = (acc + sum_r buck[r]*table_v[r]) / Z   -> ctx [b][s][h*64+d]
#pragma unroll
  for (int nt = 0; nt < 4; ++nt) {
    int d = nt * 16 + n;
#pragma unroll
    for (int r = 0; r < 8; ++r) {
      float w2 = 0.f;
#pragma unroll
      for (int j = 0; j < 9; ++j) w2 += buck[r][j] * tv[j * DHEAD + d];
      int qrow = q0 + 8 * hf + r;
      float val = (acc[nt][r] + w2) / rsum[r];
      ctx[((size_t)(b * S_LEN + qrow)) * HIDN + h * DHEAD + d] = (bf16_t)val;
    }
  }
}

// ---------------------------------------------------------------------------
extern "C" void kernel_launch(void* const* d_in, const int* in_sizes, int n_in,
                              void* d_out, int out_size, void* d_ws,
                              size_t ws_size, hipStream_t stream) {
  (void)in_sizes; (void)n_in; (void)out_size; (void)ws_size;
  const float* x  = (const float*)d_in[0];
  const float* Wq = (const float*)d_in[1];
  const float* bq = (const float*)d_in[2];
  const float* Wk = (const float*)d_in[3];
  const float* bk = (const float*)d_in[4];
  const float* Wv = (const float*)d_in[5];
  const float* bv = (const float*)d_in[6];
  const float* Wo = (const float*)d_in[7];
  const float* bo = (const float*)d_in[8];
  const float* table_k = (const float*)d_in[9];
  const float* table_v = (const float*)d_in[10];
  float* out = (float*)d_out;

  char* ws = (char*)d_ws;
  size_t off = 0;
  auto alloc = [&](size_t bytes) -> void* {
    void* p = ws + off;
    off += (bytes + 255) & ~(size_t)255;
    return p;
  };
  bf16_t* xb   = (bf16_t*)alloc((size_t)MROWS * HIDN * 2);   // 8 MB
  bf16_t* WqT  = (bf16_t*)alloc((size_t)HIDN * HIDN * 2);
  bf16_t* WkT  = (bf16_t*)alloc((size_t)HIDN * HIDN * 2);
  bf16_t* WvT  = (bf16_t*)alloc((size_t)HIDN * HIDN * 2);
  bf16_t* WoT  = (bf16_t*)alloc((size_t)HIDN * HIDN * 2);
  bf16_t* Qh   = (bf16_t*)alloc((size_t)BATCH * NHEAD * S_LEN * DHEAD * 2);
  bf16_t* Kh   = (bf16_t*)alloc((size_t)BATCH * NHEAD * S_LEN * DHEAD * 2);
  bf16_t* Vth  = (bf16_t*)alloc((size_t)BATCH * NHEAD * DHEAD * S_LEN * 2);
  float*  qrel = (float*) alloc((size_t)BATCH * NHEAD * S_LEN * 9 * 4);
  bf16_t* ctx  = (bf16_t*)alloc((size_t)MROWS * HIDN * 2);

  // 1) casts / packing
  {
    int n = MROWS * HIDN;
    cast_f32_to_bf16<<<(n + 255) / 256, 256, 0, stream>>>(x, xb, n);
    int nw = HIDN * HIDN;
    int g = (nw + 255) / 256;
    transpose_cast<<<g, 256, 0, stream>>>(Wq, WqT);
    transpose_cast<<<g, 256, 0, stream>>>(Wk, WkT);
    transpose_cast<<<g, 256, 0, stream>>>(Wv, WvT);
    transpose_cast<<<g, 256, 0, stream>>>(Wo, WoT);
  }
  // 2) Q/K/V projections (32x64 tile per wave; 4 waves per block)
  const int gemm_blocks = (MROWS / 32) * (HIDN / 64) / 4;   // 512
  gemm_wmma<<<gemm_blocks, 128, 0, stream>>>(xb, WqT, bq, Qh, MROWS, HIDN, HIDN, 0);
  gemm_wmma<<<gemm_blocks, 128, 0, stream>>>(xb, WkT, bk, Kh, MROWS, HIDN, HIDN, 0);
  gemm_wmma<<<gemm_blocks, 128, 0, stream>>>(xb, WvT, bv, Vth, MROWS, HIDN, HIDN, 1);
  // 3) content-position dot products
  {
    int n = BATCH * NHEAD * S_LEN * 9;
    qrel_kernel<<<(n + 255) / 256, 256, 0, stream>>>(Qh, table_k, qrel);
  }
  // 4) fused attention (scores + rel bias + softmax + P@V + rel-value term)
  attn_kernel<<<BATCH * NHEAD * 32, 128, 0, stream>>>(Qh, Kh, Vth, qrel,
                                                      table_v, ctx);
  // 5) output projection (f32 result)
  gemm_wmma<<<gemm_blocks, 128, 0, stream>>>(ctx, WoT, bo, out, MROWS, HIDN, HIDN, 2);
}
